// SketchRNN_47725676593304
// MI455X (gfx1250) — compile-verified, matrix operand order checked by reading, not run on hardware
//
#include <hip/hip_runtime.h>
#include <hip/hip_bf16.h>
#include <stdint.h>

// ---------------------------------------------------------------------------
// SketchRNN VAE forward for MI455X (gfx1250, wave32, WMMA).
//
// Roofline: decoder recurrence = 2.2 TFLOP of (512x2048)@(2048x8192) GEMMs.
// dec_Whh in bf16 = 33.5MB -> resident in 192MB L2 across all 128 steps, so
// the problem is compute bound => v_wmma_f32_16x16x32_bf16 with f32 accum.
// Input-side GEMVs (x@Wih.T + b) are fused into GEMM epilogues.
// ---------------------------------------------------------------------------

#define BSZ    512
#define TLEN   128
#define ZDIM   128
#define HEC    512     // encoder hidden
#define HDC    2048    // decoder hidden
#define NOUTP  123
#define DROW   258     // data row stride = (MAXLEN+1)*2

typedef __attribute__((ext_vector_type(16))) __bf16 v16bf;
typedef __attribute__((ext_vector_type(8)))  float  v8f;

union Frag { v16bf v; uint4 q[2]; };

__device__ __forceinline__ v8f wmma_bf16(v16bf a, v16bf b, v8f c) {
  // D = A(16x32) * B(32x16) + C, f32 accumulate
  return __builtin_amdgcn_wmma_f32_16x16x32_bf16(false, a, false, b,
                                                 (short)0, c, false, false);
}

__device__ __forceinline__ unsigned short f2bf(float f) {
  union { float f; unsigned u; } x; x.f = f;
  unsigned u = x.u;
  unsigned r = u + 0x7fffu + ((u >> 16) & 1u);   // round-to-nearest-even
  return (unsigned short)(r >> 16);
}
__device__ __forceinline__ float bf2f(unsigned short h) {
  union { unsigned u; float f; } x; x.u = ((unsigned)h) << 16;
  return x.f;
}
__device__ __forceinline__ float sigm(float x) { return 1.0f / (1.0f + __expf(-x)); }

// ---------------------------------------------------------------------------
// Templated WMMA GEMM:  C[M,N] = A[M,K](bf16) * W[N,K](bf16)^T  (+ epilogue)
// Wave tile: 32(M) x (NFRAG*16)(N).  Block: 4 waves along N.
// A/B fragments follow the CDNA5 16-bit 16x32 lane layout:
//   lane<16 : row = laneM, holds K[0..7] (q0) and K[16..23] (q1)
//   lane>=16: row = laneM, holds K[8..15](q0) and K[24..31] (q1)
// i.e. per-lane base offset = (lane>>4)*8 elements, +16 for the second quad.
// ---------------------------------------------------------------------------
enum { M_PLAIN = 0, M_ENCF = 1, M_ENCB = 2, M_DEC = 3, M_OUTPROJ = 4 };

template <int MODE, int NFRAG>
__global__ __launch_bounds__(128)
void wmma_gemm(const unsigned short* __restrict__ A,   // M x K bf16
               const unsigned short* __restrict__ W,   // N x K bf16
               int K, float* __restrict__ out, int ldo,
               const float* __restrict__ xdata,        // raw data tensor
               const float* __restrict__ wih,          // fp32 input weights
               const float* __restrict__ bias,
               const float* __restrict__ addmat,       // dec: precomputed z@Wz.T
               const int*   __restrict__ lengths,
               int t)
{
  const int lane   = threadIdx.x & 31;
  const int wave   = threadIdx.x >> 5;
  const int laneM  = lane & 15;
  const int laneHi = lane >> 4;                 // 0 or 1
  const int mBase  = blockIdx.x * 32;
  const int nBase  = blockIdx.y * (4 * NFRAG * 16) + wave * NFRAG * 16;

  v8f acc[NFRAG][2];
#pragma unroll
  for (int f = 0; f < NFRAG; ++f) {
    acc[f][0] = (v8f)(0.0f);
    acc[f][1] = (v8f)(0.0f);
  }

  const unsigned short* aRow0 = A + (size_t)(mBase + laneM) * K + laneHi * 8;
  const unsigned short* aRow1 = aRow0 + (size_t)16 * K;
  const unsigned short* wRow[NFRAG];
#pragma unroll
  for (int f = 0; f < NFRAG; ++f)
    wRow[f] = W + (size_t)(nBase + f * 16 + laneM) * K + laneHi * 8;

  for (int k0 = 0; k0 < K; k0 += 32) {
    Frag a0, a1, b[NFRAG];
    a0.q[0] = *(const uint4*)(aRow0 + k0);
    a0.q[1] = *(const uint4*)(aRow0 + k0 + 16);
    a1.q[0] = *(const uint4*)(aRow1 + k0);
    a1.q[1] = *(const uint4*)(aRow1 + k0 + 16);
#pragma unroll
    for (int f = 0; f < NFRAG; ++f) {
      b[f].q[0] = *(const uint4*)(wRow[f] + k0);
      b[f].q[1] = *(const uint4*)(wRow[f] + k0 + 16);
    }
    if (k0 + 64 < K) {                         // prefetch next K tile (L2-hot)
      __builtin_prefetch(aRow0 + k0 + 64, 0, 1);
      __builtin_prefetch(wRow[0] + k0 + 64, 0, 1);
    }
#pragma unroll
    for (int f = 0; f < NFRAG; ++f) {
      acc[f][0] = wmma_bf16(a0.v, b[f].v, acc[f][0]);
      acc[f][1] = wmma_bf16(a1.v, b[f].v, acc[f][1]);
    }
  }

  // ---- epilogue: C-frag layout => m = mBase + mi*16 + laneHi*8 + r,
  //                                 n = nBase + f*16 + laneM
#pragma unroll
  for (int f = 0; f < NFRAG; ++f) {
    const int n = nBase + f * 16 + laneM;
#pragma unroll
    for (int mi = 0; mi < 2; ++mi) {
#pragma unroll
      for (int r = 0; r < 8; ++r) {
        const int m = mBase + mi * 16 + laneHi * 8 + r;
        float v = acc[f][mi][r];
        if (MODE == M_PLAIN) {
          out[(size_t)m * ldo + n] = v;
        } else if (MODE == M_ENCF) {
          // enc_in[b,t] = data[b, t+1, :]
          float x0 = xdata[(size_t)m * DROW + (t + 1) * 2 + 0];
          float x1 = xdata[(size_t)m * DROW + (t + 1) * 2 + 1];
          v += x0 * wih[n * 2 + 0] + x1 * wih[n * 2 + 1] + bias[n];
          out[(size_t)m * ldo + n] = v;
        } else if (MODE == M_ENCB) {
          int L = lengths[m]; L = min(max(L, 1), TLEN);
          int idx = L - 1 - t; idx = max(idx, 0);
          float x0 = xdata[(size_t)m * DROW + (idx + 1) * 2 + 0];
          float x1 = xdata[(size_t)m * DROW + (idx + 1) * 2 + 1];
          v += x0 * wih[n * 2 + 0] + x1 * wih[n * 2 + 1] + bias[n];
          out[(size_t)m * ldo + n] = v;
        } else if (MODE == M_DEC) {
          // dec_in[b,t] = data[b, t, :]; cols 2..4 of dec_Wih hit zero pad
          float x0 = xdata[(size_t)m * DROW + t * 2 + 0];
          float x1 = xdata[(size_t)m * DROW + t * 2 + 1];
          v += addmat[(size_t)m * ldo + n]
             + x0 * wih[n * 133 + 0] + x1 * wih[n * 133 + 1] + bias[n];
          out[(size_t)m * ldo + n] = v;
        } else { // M_OUTPROJ: write params[b=m, t, n] for n < NOUT
          v += bias[n];
          if (n < NOUTP)
            out[((size_t)m * TLEN + t) * NOUTP + n] = v;
        }
      }
    }
  }
}

// ---------------------------------------------------------------------------
// Small VALU kernels
// ---------------------------------------------------------------------------
__global__ void k_f32_to_bf16(const float* __restrict__ in,
                              unsigned short* __restrict__ out, int n) {
  int i = blockIdx.x * blockDim.x + threadIdx.x;
  if (i < n) out[i] = f2bf(in[i]);
}

// dec_Wih[:, 5:133] -> bf16 (8192 x 128), K-contiguous & aligned
__global__ void k_pack_decWz(const float* __restrict__ decWih,
                             unsigned short* __restrict__ out) {
  int i = blockIdx.x * blockDim.x + threadIdx.x;
  if (i >= 4 * HDC * ZDIM) return;
  int n = i / ZDIM, k = i % ZDIM;
  out[i] = f2bf(decWih[(size_t)n * 133 + 5 + k]);
}

// out_W (123 x 2048) -> bf16 padded to 128 rows
__global__ void k_pack_outW(const float* __restrict__ outW,
                            unsigned short* __restrict__ out) {
  int i = blockIdx.x * blockDim.x + threadIdx.x;
  if (i >= 128 * HDC) return;
  int n = i / HDC, k = i % HDC;
  out[i] = (n < NOUTP) ? f2bf(outW[(size_t)n * HDC + k]) : (unsigned short)0;
}

// LSTM cell pointwise update (gate order: i, f, g, o)
template <int H, bool MASKED>
__global__ void k_lstm_cell(const float* __restrict__ gates,
                            float* __restrict__ c,
                            unsigned short* __restrict__ h,
                            const int* __restrict__ lengths, int t) {
  int i = blockIdx.x * blockDim.x + threadIdx.x;
  if (i >= BSZ * H) return;
  int b = i / H, j = i % H;
  if (MASKED) {
    int L = lengths[b]; L = min(max(L, 1), TLEN);
    if (t >= L) return;                        // keep old h,c
  }
  const float* g = gates + (size_t)b * 4 * H;
  float gi = g[j], gf = g[H + j], gg = g[2 * H + j], go = g[3 * H + j];
  float c2 = sigm(gf) * c[i] + sigm(gi) * tanhf(gg);
  float h2 = sigm(go) * tanhf(c2);
  c[i] = c2;
  h[i] = f2bf(h2);
}

// zmzl[b, 0:256] = [hF, hB] @ enc_out_W.T + enc_out_b
__global__ void k_enc_out(const unsigned short* __restrict__ hF,
                          const unsigned short* __restrict__ hB,
                          const float* __restrict__ W,   // 256 x 1024
                          const float* __restrict__ bias,
                          float* __restrict__ zmzl) {
  int i = blockIdx.x * blockDim.x + threadIdx.x;
  if (i >= BSZ * 2 * ZDIM) return;
  int b = i / (2 * ZDIM), n = i % (2 * ZDIM);
  const float* wr = W + (size_t)n * (2 * HEC);
  float acc = bias[n];
  for (int k = 0; k < HEC; ++k) acc += bf2f(hF[(size_t)b * HEC + k]) * wr[k];
  for (int k = 0; k < HEC; ++k) acc += bf2f(hB[(size_t)b * HEC + k]) * wr[HEC + k];
  zmzl[i] = acc;
}

// reparameterize; also emit zm/zl outputs
__global__ void k_latent(const float* __restrict__ zmzl,
                         const float* __restrict__ eps,
                         float* __restrict__ zm_out, float* __restrict__ zl_out,
                         float* __restrict__ zf, unsigned short* __restrict__ zbf) {
  int i = blockIdx.x * blockDim.x + threadIdx.x;
  if (i >= BSZ * ZDIM) return;
  int b = i / ZDIM, n = i % ZDIM;
  float zm = zmzl[(size_t)b * 2 * ZDIM + n];
  float zl = zmzl[(size_t)b * 2 * ZDIM + ZDIM + n];
  zm_out[i] = zm;
  zl_out[i] = zl;
  float z = zm + __expf(0.5f * zl) * eps[i];
  zf[i] = z;
  zbf[i] = f2bf(z);
}

// h0,c0 = split(tanh(z @ init_W.T + init_b))
__global__ void k_init_state(const float* __restrict__ zf,
                             const float* __restrict__ W,   // 4096 x 128
                             const float* __restrict__ bias,
                             unsigned short* __restrict__ hdec,
                             float* __restrict__ cdec) {
  int i = blockIdx.x * blockDim.x + threadIdx.x;
  if (i >= BSZ * 2 * HDC) return;
  int b = i / (2 * HDC), n = i % (2 * HDC);
  const float* wr = W + (size_t)n * ZDIM;
  const float* zr = zf + (size_t)b * ZDIM;
  float acc = bias[n];
#pragma unroll 4
  for (int k = 0; k < ZDIM; ++k) acc += zr[k] * wr[k];
  float v = tanhf(acc);
  if (n < HDC) hdec[(size_t)b * HDC + n] = f2bf(v);
  else         cdec[(size_t)b * HDC + (n - HDC)] = v;
}

// ---------------------------------------------------------------------------
extern "C" void kernel_launch(void* const* d_in, const int* in_sizes, int n_in,
                              void* d_out, int out_size, void* d_ws, size_t ws_size,
                              hipStream_t stream) {
  const float* data     = (const float*)d_in[0];
  const float* eps      = (const float*)d_in[1];
  const float* encWih_f = (const float*)d_in[2];
  const float* encWhh_f = (const float*)d_in[3];
  const float* encB_f   = (const float*)d_in[4];
  const float* encWih_b = (const float*)d_in[5];
  const float* encWhh_b = (const float*)d_in[6];
  const float* encB_b   = (const float*)d_in[7];
  const float* encOutW  = (const float*)d_in[8];
  const float* encOutB  = (const float*)d_in[9];
  const float* initW    = (const float*)d_in[10];
  const float* initB    = (const float*)d_in[11];
  const float* decWih   = (const float*)d_in[12];
  const float* decWhh   = (const float*)d_in[13];
  const float* decB     = (const float*)d_in[14];
  const float* outW     = (const float*)d_in[15];
  const float* outB     = (const float*)d_in[16];
  const int*   lengths  = (const int*)  d_in[17];

  float* params = (float*)d_out;                              // B*T*123
  float* zm_out = (float*)d_out + (size_t)BSZ * TLEN * NOUTP; // B*128
  float* zl_out = zm_out + (size_t)BSZ * ZDIM;

  // ---- workspace carve-up (~91 MB), 256B aligned slabs
  char* ws = (char*)d_ws;
  size_t off = 0;
  auto carve = [&](size_t bytes) -> void* {
    void* p = ws + off;
    off = (off + bytes + 255) & ~(size_t)255;
    return p;
  };
  unsigned short* whhf_bf  = (unsigned short*)carve((size_t)4*HEC*HEC*2);
  unsigned short* whhb_bf  = (unsigned short*)carve((size_t)4*HEC*HEC*2);
  unsigned short* dwhh_bf  = (unsigned short*)carve((size_t)4*HDC*HDC*2);
  unsigned short* dwz_bf   = (unsigned short*)carve((size_t)4*HDC*ZDIM*2);
  unsigned short* outw_bf  = (unsigned short*)carve((size_t)128*HDC*2);
  unsigned short* h_f      = (unsigned short*)carve((size_t)BSZ*HEC*2);
  unsigned short* h_b      = (unsigned short*)carve((size_t)BSZ*HEC*2);
  float*          c_f      = (float*)carve((size_t)BSZ*HEC*4);
  float*          c_b      = (float*)carve((size_t)BSZ*HEC*4);
  float*          gates_f  = (float*)carve((size_t)BSZ*4*HEC*4);
  float*          gates_b  = (float*)carve((size_t)BSZ*4*HEC*4);
  unsigned short* h_dec    = (unsigned short*)carve((size_t)BSZ*HDC*2);
  float*          c_dec    = (float*)carve((size_t)BSZ*HDC*4);
  float*          gates_d  = (float*)carve((size_t)BSZ*4*HDC*4);
  float*          dxz      = (float*)carve((size_t)BSZ*4*HDC*4);
  float*          zmzl     = (float*)carve((size_t)BSZ*2*ZDIM*4);
  float*          zf       = (float*)carve((size_t)BSZ*ZDIM*4);
  unsigned short* zbf      = (unsigned short*)carve((size_t)BSZ*ZDIM*2);
  (void)ws_size; (void)n_in; (void)in_sizes; (void)out_size;

  const int TB = 256;
  auto blks = [](size_t n, int tb) { return (unsigned)((n + tb - 1) / tb); };

  // ---- weight conversion / packing
  k_f32_to_bf16<<<blks((size_t)4*HEC*HEC, TB), TB, 0, stream>>>(encWhh_f, whhf_bf, 4*HEC*HEC);
  k_f32_to_bf16<<<blks((size_t)4*HEC*HEC, TB), TB, 0, stream>>>(encWhh_b, whhb_bf, 4*HEC*HEC);
  k_f32_to_bf16<<<blks((size_t)4*HDC*HDC, TB), TB, 0, stream>>>(decWhh,  dwhh_bf, 4*HDC*HDC);
  k_pack_decWz <<<blks((size_t)4*HDC*ZDIM, TB), TB, 0, stream>>>(decWih, dwz_bf);
  k_pack_outW  <<<blks((size_t)128*HDC,   TB), TB, 0, stream>>>(outW, outw_bf);

  // ---- zero initial encoder state (deterministic every call)
  hipMemsetAsync(h_f, 0, (size_t)BSZ*HEC*2, stream);
  hipMemsetAsync(h_b, 0, (size_t)BSZ*HEC*2, stream);
  hipMemsetAsync(c_f, 0, (size_t)BSZ*HEC*4, stream);
  hipMemsetAsync(c_b, 0, (size_t)BSZ*HEC*4, stream);

  // ---- bidirectional encoder: per-step WMMA gates GEMM + masked cell
  {
    dim3 grid(BSZ/32, (4*HEC)/256), blk(128);
    for (int t = 0; t < TLEN; ++t) {
      wmma_gemm<M_ENCF,4><<<grid, blk, 0, stream>>>(h_f, whhf_bf, HEC, gates_f, 4*HEC,
          data, encWih_f, encB_f, nullptr, lengths, t);
      wmma_gemm<M_ENCB,4><<<grid, blk, 0, stream>>>(h_b, whhb_bf, HEC, gates_b, 4*HEC,
          data, encWih_b, encB_b, nullptr, lengths, t);
      k_lstm_cell<HEC,true><<<blks((size_t)BSZ*HEC, TB), TB, 0, stream>>>(gates_f, c_f, h_f, lengths, t);
      k_lstm_cell<HEC,true><<<blks((size_t)BSZ*HEC, TB), TB, 0, stream>>>(gates_b, c_b, h_b, lengths, t);
    }
  }

  // ---- latent path
  k_enc_out<<<blks((size_t)BSZ*2*ZDIM, TB), TB, 0, stream>>>(h_f, h_b, encOutW, encOutB, zmzl);
  k_latent <<<blks((size_t)BSZ*ZDIM,  TB), TB, 0, stream>>>(zmzl, eps, zm_out, zl_out, zf, zbf);
  k_init_state<<<blks((size_t)BSZ*2*HDC, TB), TB, 0, stream>>>(zf, initW, initB, h_dec, c_dec);

  // ---- z @ Wz.T once (constant over t)
  {
    dim3 grid(BSZ/32, (4*HDC)/256), blk(128);
    wmma_gemm<M_PLAIN,4><<<grid, blk, 0, stream>>>(zbf, dwz_bf, ZDIM, dxz, 4*HDC,
        nullptr, nullptr, nullptr, nullptr, nullptr, 0);
  }

  // ---- decoder: gates GEMM -> cell -> fused per-step output projection
  {
    dim3 gridG(BSZ/32, (4*HDC)/256), blk(128);
    dim3 gridO(BSZ/32, 1);
    for (int t = 0; t < TLEN; ++t) {
      wmma_gemm<M_DEC,4><<<gridG, blk, 0, stream>>>(h_dec, dwhh_bf, HDC, gates_d, 4*HDC,
          data, decWih, decB, dxz, nullptr, t);
      k_lstm_cell<HDC,false><<<blks((size_t)BSZ*HDC, TB), TB, 0, stream>>>(gates_d, c_dec, h_dec, nullptr, t);
      wmma_gemm<M_OUTPROJ,2><<<gridO, blk, 0, stream>>>(h_dec, outw_bf, HDC, params, NOUTP,
          nullptr, nullptr, outB, nullptr, nullptr, t);
    }
  }
}
